// TransformerEncoderLayer_79998060855388
// MI455X (gfx1250) — matrix operand order repaired and checked by reading, not compiled, not run on hardware
//
#include <hip/hip_runtime.h>

#define Bb   4
#define Ss   2048
#define Ee   1024
#define Hn   16
#define Dd   64
#define HIDc 4096

typedef __bf16 bf16_t;
typedef __attribute__((ext_vector_type(16))) __bf16 v16bf;
typedef __attribute__((ext_vector_type(8)))  __bf16 v8bf;
typedef __attribute__((ext_vector_type(4)))  __bf16 v4bf;
typedef __attribute__((ext_vector_type(8)))  float  v8f;

// ---------------------------------------------------------------------------
// WMMA helpers (CDNA5 16x16x32 bf16 -> f32). Layouts per CDNA5 ISA:
//  A 16x32: lane m=L&15, half=L>>4; elem e=2g+p -> K=(g>>2)*16 + half*8 + (g&3)*2 + p
//  B 32x16: lane n=L&15, half=L>>4; elem e=2g+p -> K=(g>>2)*16 + half*8 + (g&3)*2 + p
//  C 16x16: vgpr r, lane<16 -> (M=r, N=lane); lane>=16 -> (M=8+r, N=lane-16)
// ---------------------------------------------------------------------------
static __device__ __forceinline__ v8f wmma_bf(v16bf a, v16bf b, v8f c) {
  return __builtin_amdgcn_wmma_f32_16x16x32_bf16(false, a, false, b, (short)0, c,
                                                 false, false);
}

// A tile (16x32) from row-major source, K contiguous.
static __device__ __forceinline__ v16bf load_a_rm(const bf16_t* __restrict__ p, long ld) {
  const int lane = threadIdx.x & 31;
  const int m = lane & 15, half = lane >> 4;
  const bf16_t* q = p + (long)m * ld + (half << 3);
  v16bf a;
#pragma unroll
  for (int g = 0; g < 8; ++g) {
    const int off = ((g >> 2) << 4) + ((g & 3) << 1);
    a[2 * g]     = q[off];
    a[2 * g + 1] = q[off + 1];
  }
  return a;
}

// B tile (32x16) where element (k,n) lives at p[n*ld + k] (K contiguous per column).
static __device__ __forceinline__ v16bf load_b_nm(const bf16_t* __restrict__ p, long ld) {
  const int lane = threadIdx.x & 31;
  const int n = lane & 15, half = lane >> 4;
  const bf16_t* q = p + (long)n * ld + (half << 3);
  v16bf b;
#pragma unroll
  for (int g = 0; g < 8; ++g) {
    const int off = ((g >> 2) << 4) + ((g & 3) << 1);
    b[2 * g]     = q[off];
    b[2 * g + 1] = q[off + 1];
  }
  return b;
}

// ---------------------------------------------------------------------------
// 0) Weight repack to bf16. conv weights (Co,Ci,3) -> (3,Co,Ci) so A-tile loads
//    are contiguous along Ci.
// ---------------------------------------------------------------------------
__global__ __launch_bounds__(256) void pack_weights_kernel(
    const float* __restrict__ Wq, const float* __restrict__ Wk,
    const float* __restrict__ Wv, const float* __restrict__ Wo,
    const float* __restrict__ c1w, const float* __restrict__ c2w,
    bf16_t* __restrict__ Wqb, bf16_t* __restrict__ Wkb, bf16_t* __restrict__ Wvb,
    bf16_t* __restrict__ Wob, bf16_t* __restrict__ W1p, bf16_t* __restrict__ W2p) {
  const long i = (long)blockIdx.x * 256 + threadIdx.x;
  const long stride = (long)gridDim.x * 256;
  for (long t = i; t < (long)Dd * Dd; t += stride) {
    Wqb[t] = (bf16_t)Wq[t];
    Wkb[t] = (bf16_t)Wk[t];
    Wvb[t] = (bf16_t)Wv[t];
  }
  for (long t = i; t < (long)Ee * Ee; t += stride) Wob[t] = (bf16_t)Wo[t];
  for (long t = i; t < (long)2 * HIDc * Ee * 3; t += stride) {
    const long tap = t % 3;
    const long ci  = (t / 3) % Ee;
    const long co  = t / (3L * Ee);
    W1p[(tap * 2 * HIDc + co) * (long)Ee + ci] = (bf16_t)c1w[t];
  }
  for (long t = i; t < (long)Ee * HIDc * 3; t += stride) {
    const long tap = t % 3;
    const long ci  = (t / 3) % HIDc;
    const long co  = t / (3L * HIDc);
    W2p[(tap * (long)Ee + co) * (long)HIDc + ci] = (bf16_t)c2w[t];
  }
}

// ---------------------------------------------------------------------------
// 1) LayerNorm over E=1024 (one 256-thread block per row), bf16 out,
//    optional per-row mask multiply (used for the FFN input).
// ---------------------------------------------------------------------------
__global__ __launch_bounds__(256) void ln_bf16_kernel(
    const float* __restrict__ x, const float* __restrict__ gamma,
    const float* __restrict__ beta, const int* __restrict__ mask,
    bf16_t* __restrict__ out) {
  const long row = blockIdx.x;
  const float* xr = x + row * (long)Ee;
  float v[4], s = 0.f, ss = 0.f;
#pragma unroll
  for (int i = 0; i < 4; ++i) {
    v[i] = xr[threadIdx.x + (i << 8)];
    s += v[i];
    ss += v[i] * v[i];
  }
#pragma unroll
  for (int off = 16; off; off >>= 1) {
    s  += __shfl_xor(s, off, 32);
    ss += __shfl_xor(ss, off, 32);
  }
  __shared__ float red[16];
  const int wid = threadIdx.x >> 5;
  if ((threadIdx.x & 31) == 0) { red[wid] = s; red[8 + wid] = ss; }
  __syncthreads();
  float st = 0.f, sst = 0.f;
#pragma unroll
  for (int i = 0; i < 8; ++i) { st += red[i]; sst += red[8 + i]; }
  const float mu   = st * (1.0f / Ee);
  const float var  = sst * (1.0f / Ee) - mu * mu;
  const float rstd = rsqrtf(var + 1e-5f);
  const float mm   = mask ? ((mask[row] != 0) ? 1.0f : 0.0f) : 1.0f;
  bf16_t* orow = out + row * (long)Ee;
#pragma unroll
  for (int i = 0; i < 4; ++i) {
    const int c = threadIdx.x + (i << 8);
    orow[c] = (bf16_t)(((v[i] - mu) * rstd * gamma[c] + beta[c]) * mm);
  }
}

// ---------------------------------------------------------------------------
// 2) Per-head projection: X(B*S*H, 64) @ W^T(64,64) -> Y bf16. Wave = 16x64 tile.
//    A 16-row m-tile is exactly the 16 heads of one (b,s) token.
// ---------------------------------------------------------------------------
__global__ __launch_bounds__(256) void proj_head_kernel(
    const bf16_t* __restrict__ X, const bf16_t* __restrict__ W,
    bf16_t* __restrict__ Y) {
  const int wave = threadIdx.x >> 5;
  const int lane = threadIdx.x & 31;
  const long mt = (long)blockIdx.x * 8 + wave;
  const bf16_t* a0 = X + mt * 16 * Dd;
  v8f acc[4] = {};
#pragma unroll
  for (int kc = 0; kc < 2; ++kc) {
    const v16bf a = load_a_rm(a0 + kc * 32, Dd);
#pragma unroll
    for (int nt = 0; nt < 4; ++nt)
      acc[nt] = wmma_bf(a, load_b_nm(W + (long)(nt * 16) * Dd + kc * 32, Dd), acc[nt]);
  }
  const int half = lane >> 4, col = lane & 15;
#pragma unroll
  for (int nt = 0; nt < 4; ++nt)
#pragma unroll
    for (int r = 0; r < 8; ++r)
      Y[(mt * 16 + half * 8 + r) * Dd + nt * 16 + col] = (bf16_t)acc[nt][r];
}

// Same GEMM, but writes V transposed: Yt[((b*H + h)*D + d)*S + s] so the
// attention P@V B-operand is K(=s)-contiguous.
__global__ __launch_bounds__(256) void proj_head_tr_kernel(
    const bf16_t* __restrict__ X, const bf16_t* __restrict__ W,
    bf16_t* __restrict__ Yt) {
  const int wave = threadIdx.x >> 5;
  const int lane = threadIdx.x & 31;
  const long mt = (long)blockIdx.x * 8 + wave;  // == flat (b*S + s)
  const bf16_t* a0 = X + mt * 16 * Dd;
  v8f acc[4] = {};
#pragma unroll
  for (int kc = 0; kc < 2; ++kc) {
    const v16bf a = load_a_rm(a0 + kc * 32, Dd);
#pragma unroll
    for (int nt = 0; nt < 4; ++nt)
      acc[nt] = wmma_bf(a, load_b_nm(W + (long)(nt * 16) * Dd + kc * 32, Dd), acc[nt]);
  }
  const int half = lane >> 4, col = lane & 15;
  const long bidx = mt / Ss, s = mt % Ss;
#pragma unroll
  for (int nt = 0; nt < 4; ++nt)
#pragma unroll
    for (int r = 0; r < 8; ++r) {
      const int hh = half * 8 + r;  // head index within this token's 16 rows
      Yt[((bidx * Hn + hh) * (long)Dd + nt * 16 + col) * Ss + s] = (bf16_t)acc[nt][r];
    }
}

// ---------------------------------------------------------------------------
// 3) Flash attention. 8 waves/block share (b,h): K/V chunk tiles staged in
//    double-buffered LDS (stage kt+1 under compute of kt, one barrier/iter).
//    ALiBi + mask + 1/sqrt(E). P re-layout C->A via per-wave LDS tile.
// ---------------------------------------------------------------------------
#define KLD 72  // kbuf row stride (elems): 16B-aligned rows, conflict-free reads
#define VLD 56  // vbuf row stride
#define PLD 36  // pbuf row stride

__global__ __launch_bounds__(256) void attn_kernel(
    const bf16_t* __restrict__ Q, const bf16_t* __restrict__ K,
    const bf16_t* __restrict__ Vt, const int* __restrict__ mask,
    bf16_t* __restrict__ O) {
  __shared__ bf16_t pbuf[8][16 * PLD];
  __shared__ bf16_t kbuf[2][32 * KLD];  // [j 0..31][d 0..63]
  __shared__ bf16_t vbuf[2][64 * VLD];  // [d 0..63][l 0..31]
  const int tid  = threadIdx.x;
  const int wave = tid >> 5;
  const int lane = tid & 31;
  const int half = lane >> 4;
  const int col  = lane & 15;
  const int b = blockIdx.y >> 4;  // H = 16
  const int h = blockIdx.y & 15;
  const int qt = blockIdx.x * 8 + wave;

  const bf16_t* Kbase = K + (long)b * Ss * Ee + h * Dd;     // row j, ld=Ee
  const bf16_t* Vbase = Vt + (long)(b * Hn + h) * Dd * Ss;  // row d, ld=Ss

  const bf16_t* qp = Q + (long)(b * Ss + qt * 16) * Ee + h * Dd;
  const v16bf qa0 = load_a_rm(qp, Ee);
  const v16bf qa1 = load_a_rm(qp + 32, Ee);

  const float inv = 0.03125f;  // 1/sqrt(E)
  const float sl  = exp2f(-(float)(h + 1)) * inv;
  float mx[8], ls[8];
#pragma unroll
  for (int r = 0; r < 8; ++r) { mx[r] = -3.0e38f; ls[r] = 0.0f; }
  v8f acc[4] = {};
  const int i0 = qt * 16 + half * 8;
  bf16_t* pb = &pbuf[wave][0];
  const int kj = tid >> 3, kq = tid & 7;  // K staging: 32 rows x 8 x (8 elems)

  auto stage_kv = [&](int kt, int sel) {
    *(v8bf*)&kbuf[sel][kj * KLD + kq * 8] =
        *(const v8bf*)(Kbase + (long)(kt * 32 + kj) * Ee + kq * 8);
    for (int j = tid; j < 512; j += 256) {  // 64 rows x 8 x (4 elems)
      const int d = j >> 3, q = j & 7;
      *(v4bf*)&vbuf[sel][d * VLD + q * 4] =
          *(const v4bf*)(Vbase + (long)d * Ss + kt * 32 + q * 4);
    }
  };

  stage_kv(0, 0);
  __syncthreads();

  for (int kt = 0; kt < Ss / 32; ++kt) {
    const int cur = kt & 1;
    if (kt + 1 < Ss / 32) stage_kv(kt + 1, cur ^ 1);  // overlap with compute

    float sv[2][8];
#pragma unroll
    for (int sub = 0; sub < 2; ++sub) {
      v8f sc = {};
      sc = wmma_bf(qa0, load_b_nm(&kbuf[cur][(sub * 16) * KLD], KLD), sc);
      sc = wmma_bf(qa1, load_b_nm(&kbuf[cur][(sub * 16) * KLD + 32], KLD), sc);
      const int j = kt * 32 + sub * 16 + col;
      const bool masked = (mask[b * Ss + j] == 0);
#pragma unroll
      for (int r = 0; r < 8; ++r) {
        const float xval = sc[r] * inv - fabsf((float)(i0 + r - j)) * sl;
        sv[sub][r] = masked ? -312.5f : xval;  // NEG / sqrt(E)
      }
    }
    // online softmax update over this 32-key chunk
#pragma unroll
    for (int r = 0; r < 8; ++r) {
      float m0 = fmaxf(sv[0][r], sv[1][r]);
#pragma unroll
      for (int off = 1; off < 16; off <<= 1) m0 = fmaxf(m0, __shfl_xor(m0, off, 32));
      const float mnew  = fmaxf(mx[r], m0);
      const float alpha = __expf(mx[r] - mnew);
      const float p0 = __expf(sv[0][r] - mnew);
      const float p1 = __expf(sv[1][r] - mnew);
      float rs = p0 + p1;
#pragma unroll
      for (int off = 1; off < 16; off <<= 1) rs += __shfl_xor(rs, off, 32);
      ls[r] = ls[r] * alpha + rs;
      mx[r] = mnew;
#pragma unroll
      for (int nt = 0; nt < 4; ++nt) acc[nt][r] *= alpha;
      pb[(half * 8 + r) * PLD + col]      = (bf16_t)p0;
      pb[(half * 8 + r) * PLD + 16 + col] = (bf16_t)p1;
    }
    __asm__ volatile("s_wait_dscnt 0" ::: "memory");  // same-wave P turnaround
    const v16bf pa = load_a_rm(pb, PLD);
#pragma unroll
    for (int nt = 0; nt < 4; ++nt)
      acc[nt] = wmma_bf(pa, load_b_nm(&vbuf[cur][(nt * 16) * VLD], VLD), acc[nt]);
    __syncthreads();  // staged kt+1 ready; cur free for kt+2 staging
  }

  bf16_t* op = O + (long)(b * Ss + qt * 16) * Ee + h * Dd;
#pragma unroll
  for (int nt = 0; nt < 4; ++nt)
#pragma unroll
    for (int r = 0; r < 8; ++r)
      op[(long)(half * 8 + r) * Ee + nt * 16 + col] = (bf16_t)(acc[nt][r] / ls[r]);
}

// ---------------------------------------------------------------------------
// 4) Output projection: attn(B*S,E) @ Wo^T + bo + query -> x (f32).
//    Wo column tile is block-shared -> double-buffered LDS staging.
// ---------------------------------------------------------------------------
#define BLD 48  // staged B-tile row stride (16B-aligned rows)

__global__ __launch_bounds__(256) void outproj_kernel(
    const bf16_t* __restrict__ A, const bf16_t* __restrict__ W,
    const float* __restrict__ bo, const float* __restrict__ res,
    float* __restrict__ X) {
  __shared__ bf16_t wbuf[2][64 * BLD];  // [n 0..63][k 0..31]
  const int tid = threadIdx.x;
  const int wave = tid >> 5;
  const int lane = tid & 31;
  const long mt = (long)blockIdx.x * 8 + wave;
  const int nb = blockIdx.y * 64;
  const bf16_t* a0 = A + mt * 16 * (long)Ee;
  const int sn = tid >> 2, sq = tid & 3;  // 64 rows x 4 x (8 elems)
  v8f acc[4] = {};

  auto stage = [&](int kc, int sel) {
    *(v8bf*)&wbuf[sel][sn * BLD + sq * 8] =
        *(const v8bf*)(W + (long)(nb + sn) * Ee + kc * 32 + sq * 8);
  };

  stage(0, 0);
  __syncthreads();
  for (int kc = 0; kc < Ee / 32; ++kc) {
    const int cur = kc & 1;
    if (kc + 1 < Ee / 32) stage(kc + 1, cur ^ 1);
    const v16bf a = load_a_rm(a0 + kc * 32, Ee);
#pragma unroll
    for (int nt = 0; nt < 4; ++nt)
      acc[nt] = wmma_bf(a, load_b_nm(&wbuf[cur][(nt * 16) * BLD], BLD), acc[nt]);
    __syncthreads();
  }
  const int half = lane >> 4, col = lane & 15;
#pragma unroll
  for (int nt = 0; nt < 4; ++nt) {
    const int n = nb + nt * 16 + col;
#pragma unroll
    for (int r = 0; r < 8; ++r) {
      const long m = mt * 16 + half * 8 + r;
      X[m * Ee + n] = acc[nt][r] + bo[n] + res[m * Ee + n];
    }
  }
}

// ---------------------------------------------------------------------------
// 5) conv1 (K=3 same-pad) as 3 shifted GEMMs + fused SwiGLU + mask.
//    66-column halo tile (block-shared) in double-buffered LDS; one staging
//    serves all 8 waves x 3 taps (24x traffic cut).
// ---------------------------------------------------------------------------
__global__ __launch_bounds__(256) void conv1_swiglu_kernel(
    const bf16_t* __restrict__ Xn, const bf16_t* __restrict__ W1p,
    const float* __restrict__ c1b, const int* __restrict__ mask,
    bf16_t* __restrict__ Ho) {
  __shared__ bf16_t bbuf[2][66 * BLD];  // [scol 0..65][k 0..31], scol -> s0-1+scol
  const int tid = threadIdx.x;
  const int wave = tid >> 5;
  const int lane = tid & 31;
  const int cot = blockIdx.x * 8 + wave;
  const int b  = blockIdx.y >> 5;  // S/64 = 32 s-chunks per batch
  const int s0 = (blockIdx.y & 31) << 6;
  const int colc = lane & 15;
  const bf16_t* w1b = W1p + (long)cot * 16 * Ee;
  v8f acc1[4] = {}, acc2[4] = {};

  auto stage = [&](int kc, int sel) {
    for (int j = tid; j < 264; j += 256) {  // 66 rows x 4 x (8 elems)
      const int jr = j >> 2, q = j & 3;
      const int s = s0 + jr - 1;
      v8bf val = {};
      if (s >= 0 && s < Ss)
        val = *(const v8bf*)(Xn + ((long)b * Ss + s) * Ee + kc * 32 + q * 8);
      *(v8bf*)&bbuf[sel][jr * BLD + q * 8] = val;
    }
  };

  stage(0, 0);
  __syncthreads();
  for (int kc = 0; kc < Ee / 32; ++kc) {
    const int cur = kc & 1;
    if (kc + 1 < Ee / 32) {
      stage(kc + 1, cur ^ 1);
      __builtin_prefetch(w1b + kc * 32 + 32, 0, 1);  // next weight chunk
      __builtin_prefetch(w1b + (long)HIDc * Ee + kc * 32 + 32, 0, 1);
    }
#pragma unroll
    for (int tap = 0; tap < 3; ++tap) {
      const bf16_t* w1 = w1b + (long)tap * 2 * HIDc * Ee + kc * 32;
      const v16bf a1 = load_a_rm(w1, Ee);
      const v16bf a2 = load_a_rm(w1 + (long)HIDc * Ee, Ee);
#pragma unroll
      for (int nt = 0; nt < 4; ++nt) {
        const v16bf bm = load_b_nm(&bbuf[cur][(nt * 16 + tap) * BLD], BLD);
        acc1[nt] = wmma_bf(a1, bm, acc1[nt]);
        acc2[nt] = wmma_bf(a2, bm, acc2[nt]);
      }
    }
    __syncthreads();
  }
  const int half = lane >> 4;
#pragma unroll
  for (int nt = 0; nt < 4; ++nt) {
    const int s = s0 + nt * 16 + colc;
    const float mm = (mask[b * Ss + s] != 0) ? 1.0f : 0.0f;
    bf16_t* ho = Ho + ((long)b * Ss + s) * HIDc + cot * 16 + half * 8;
#pragma unroll
    for (int r = 0; r < 8; ++r) {
      const int co = cot * 16 + half * 8 + r;
      const float x1 = acc1[nt][r] + c1b[co];
      const float x2 = acc2[nt][r] + c1b[co + HIDc];
      const float sig = 1.0f / (1.0f + __expf(-x1));
      ho[r] = (bf16_t)(x1 * sig * x2 * mm);
    }
  }
}

// ---------------------------------------------------------------------------
// 6) conv2 (K=3 same-pad) + bias + residual with original query -> output f32
// ---------------------------------------------------------------------------
__global__ __launch_bounds__(256) void conv2_res_kernel(
    const bf16_t* __restrict__ Hi, const bf16_t* __restrict__ W2p,
    const float* __restrict__ c2b, const float* __restrict__ res,
    float* __restrict__ out) {
  __shared__ bf16_t bbuf[2][66 * BLD];
  const int tid = threadIdx.x;
  const int wave = tid >> 5;
  const int lane = tid & 31;
  const int cot = blockIdx.x * 8 + wave;
  const int b  = blockIdx.y >> 5;
  const int s0 = (blockIdx.y & 31) << 6;
  const int colc = lane & 15;
  const bf16_t* w2b = W2p + (long)cot * 16 * HIDc;
  v8f acc[4] = {};

  auto stage = [&](int kc, int sel) {
    for (int j = tid; j < 264; j += 256) {
      const int jr = j >> 2, q = j & 3;
      const int s = s0 + jr - 1;
      v8bf val = {};
      if (s >= 0 && s < Ss)
        val = *(const v8bf*)(Hi + ((long)b * Ss + s) * HIDc + kc * 32 + q * 8);
      *(v8bf*)&bbuf[sel][jr * BLD + q * 8] = val;
    }
  };

  stage(0, 0);
  __syncthreads();
  for (int kc = 0; kc < HIDc / 32; ++kc) {
    const int cur = kc & 1;
    if (kc + 1 < HIDc / 32) {
      stage(kc + 1, cur ^ 1);
      __builtin_prefetch(w2b + kc * 32 + 32, 0, 1);
    }
#pragma unroll
    for (int tap = 0; tap < 3; ++tap) {
      const v16bf a = load_a_rm(w2b + (long)tap * Ee * HIDc + kc * 32, HIDc);
#pragma unroll
      for (int nt = 0; nt < 4; ++nt)
        acc[nt] = wmma_bf(a, load_b_nm(&bbuf[cur][(nt * 16 + tap) * BLD], BLD), acc[nt]);
    }
    __syncthreads();
  }
  const int half = lane >> 4;
#pragma unroll
  for (int nt = 0; nt < 4; ++nt) {
    const int s = s0 + nt * 16 + colc;
    float* op = out + ((long)b * Ss + s) * Ee + cot * 16 + half * 8;
    const float* rp = res + ((long)b * Ss + s) * Ee + cot * 16 + half * 8;
#pragma unroll
    for (int r = 0; r < 8; ++r)
      op[r] = acc[nt][r] + c2b[cot * 16 + half * 8 + r] + rp[r];
  }
}

// ---------------------------------------------------------------------------
extern "C" void kernel_launch(void* const* d_in, const int* in_sizes, int n_in,
                              void* d_out, int out_size, void* d_ws, size_t ws_size,
                              hipStream_t stream) {
  (void)in_sizes; (void)n_in; (void)out_size; (void)ws_size;
  const float* value = (const float*)d_in[0];
  const float* key_t = (const float*)d_in[1];
  const float* query = (const float*)d_in[2];
  const int*   mask  = (const int*)d_in[3];
  const float* ln1_g = (const float*)d_in[4];
  const float* ln1_b = (const float*)d_in[5];
  const float* ln2_g = (const float*)d_in[6];
  const float* ln2_b = (const float*)d_in[7];
  const float* Wq    = (const float*)d_in[8];
  const float* Wk    = (const float*)d_in[9];
  const float* Wv    = (const float*)d_in[10];
  const float* Wo    = (const float*)d_in[11];
  const float* bo    = (const float*)d_in[12];
  const float* c1w   = (const float*)d_in[13];
  const float* c1b   = (const float*)d_in[14];
  const float* c2w   = (const float*)d_in[15];
  const float* c2b   = (const float*)d_in[16];
  float* out = (float*)d_out;

  char* wp = (char*)d_ws;
  auto alloc = [&](size_t bytes) -> char* {
    char* p = wp;
    wp += (bytes + 255) & ~(size_t)255;
    return p;
  };
  const size_t NE = (size_t)Bb * Ss * Ee;  // 8,388,608
  bf16_t* qn  = (bf16_t*)alloc(NE * 2);
  bf16_t* kn  = (bf16_t*)alloc(NE * 2);
  bf16_t* vn  = (bf16_t*)alloc(NE * 2);
  bf16_t* qpp = (bf16_t*)alloc(NE * 2);
  bf16_t* kpp = (bf16_t*)alloc(NE * 2);
  bf16_t* vtp = (bf16_t*)alloc(NE * 2);    // V projected, transposed (B,H,D,S)
  bf16_t* ao  = (bf16_t*)alloc(NE * 2);
  float*  xbuf = (float*)alloc(NE * 4);
  bf16_t* xn2 = (bf16_t*)alloc(NE * 2);
  bf16_t* hbuf = (bf16_t*)alloc((size_t)Bb * Ss * HIDc * 2);
  bf16_t* Wqb = (bf16_t*)alloc((size_t)Dd * Dd * 2);
  bf16_t* Wkb = (bf16_t*)alloc((size_t)Dd * Dd * 2);
  bf16_t* Wvb = (bf16_t*)alloc((size_t)Dd * Dd * 2);
  bf16_t* Wob = (bf16_t*)alloc((size_t)Ee * Ee * 2);
  bf16_t* W1p = (bf16_t*)alloc((size_t)3 * 2 * HIDc * Ee * 2);
  bf16_t* W2p = (bf16_t*)alloc((size_t)3 * Ee * HIDc * 2);

  pack_weights_kernel<<<2048, 256, 0, stream>>>(Wq, Wk, Wv, Wo, c1w, c2w,
                                                Wqb, Wkb, Wvb, Wob, W1p, W2p);
  const int rows = Bb * Ss;
  ln_bf16_kernel<<<rows, 256, 0, stream>>>(query, ln1_g, ln1_b, nullptr, qn);
  ln_bf16_kernel<<<rows, 256, 0, stream>>>(key_t, ln1_g, ln1_b, nullptr, kn);
  ln_bf16_kernel<<<rows, 256, 0, stream>>>(value, ln1_g, ln1_b, nullptr, vn);

  proj_head_kernel<<<1024, 256, 0, stream>>>(qn, Wqb, qpp);
  proj_head_kernel<<<1024, 256, 0, stream>>>(kn, Wkb, kpp);
  proj_head_tr_kernel<<<1024, 256, 0, stream>>>(vn, Wvb, vtp);

  attn_kernel<<<dim3(Ss / 16 / 8, Bb * Hn), 256, 0, stream>>>(qpp, kpp, vtp, mask, ao);

  outproj_kernel<<<dim3(rows / 16 / 8, Ee / 64), 256, 0, stream>>>(ao, Wob, bo, query, xbuf);

  ln_bf16_kernel<<<rows, 256, 0, stream>>>(xbuf, ln2_g, ln2_b, mask, xn2);

  conv1_swiglu_kernel<<<dim3(HIDc / 16 / 8, Bb * (Ss / 64)), 256, 0, stream>>>(
      xn2, W1p, c1b, mask, hbuf);
  conv2_res_kernel<<<dim3(Ee / 16 / 8, Bb * (Ss / 64)), 256, 0, stream>>>(
      hbuf, W2p, c2b, query, out);
}
